// SparseMLP_377957122239
// MI455X (gfx1250) — matrix operand order repaired and checked by reading, not compiled, not run on hardware
//
#include <hip/hip_runtime.h>
#include <hip/hip_bf16.h>
#include <math.h>

// Problem constants (match reference)
#define NEXP 8
#define HID  1024
#define FFN  4096
#define TPE  1024   // tokens per expert

// Tiling
#define BK   32
#define LSTR 34     // LDS row stride in bf16 elements (68B = 17 banks; conflict-free)

typedef __attribute__((ext_vector_type(16))) __bf16 v16bf;
typedef __attribute__((ext_vector_type(8)))  float  v8f;

union FragBF { v16bf v; unsigned int u[8]; };

// Native bf16 converts (RNE) — let clang emit v_cvt*_bf16_f32 instead of
// the 3-VALU software round-to-nearest-even sequence.
__device__ __forceinline__ unsigned short f32_to_bf16(float f) {
  return __builtin_bit_cast(unsigned short, (__bf16)f);
}
__device__ __forceinline__ unsigned int pack2_bf16(float a, float b) {
  return (unsigned int)f32_to_bf16(a) | ((unsigned int)f32_to_bf16(b) << 16);
}
// tanh-approx GELU, branch-free: 0.5*(1+tanh(a*u)) == sigmoid(2a*u)
__device__ __forceinline__ float gelu_tanh(float x) {
  float u = x + 0.044715f * x * x * x;
  float t = __expf(-1.5957691216057308f * u);   // 2*sqrt(2/pi)
  return x / (1.0f + t);
}

// A fragment: 16x32 bf16. Lane l16 = row M; half selects K-block.
// VGPR v: K = 2v + 16*half  (pairs contiguous)
__device__ __forceinline__ v16bf load_a_frag(const unsigned short* Als,
                                             int rowBase, int l16, int half) {
  FragBF f;
#pragma unroll
  for (int v = 0; v < 8; ++v) {
    int k = 2 * v + 16 * half;
    f.u[v] = *(const unsigned int*)&Als[(rowBase + l16) * LSTR + k];
  }
  return f.v;
}
// B fragment: 32x16 bf16. Lane l16 = col N; VGPR v: K = 2v + 16*half
__device__ __forceinline__ v16bf load_b_frag(const unsigned short* Bls,
                                             int colBase, int l16, int half) {
  FragBF f;
#pragma unroll
  for (int v = 0; v < 8; ++v) {
    int k = 2 * v + 16 * half;
    f.u[v] = *(const unsigned int*)&Bls[(colBase + l16) * LSTR + k];
  }
  return f.v;
}

// ---------------- Kernel 1: h = gelu(x @ w1^T), h stored bf16 in workspace ---
// Block tile 128x256, 8 waves in 2x4 grid, wave tile 64x64 (4x4 WMMA tiles).
#define K1_BM 128
#define K1_BN 256
__global__ __launch_bounds__(256)
void moe_gemm1_gelu(const float* __restrict__ x, const float* __restrict__ w1,
                    unsigned short* __restrict__ h) {
  __shared__ unsigned short Als[2][K1_BM * LSTR];
  __shared__ unsigned short Bls[2][K1_BN * LSTR];

  const int tid = threadIdx.x;
  const int e   = blockIdx.z;
  const int tm  = blockIdx.y;   // token tile within expert
  const int tn  = blockIdx.x;   // ffn tile

  const float* Ag = x  + (size_t)(e * TPE + tm * K1_BM) * HID;
  const float* Bg = w1 + (size_t)(e * FFN + tn * K1_BN) * HID;

  const int wid  = tid >> 5;
  const int lane = tid & 31;
  const int wm   = wid >> 2;    // 0..1 (64 rows each)
  const int wn   = wid & 3;     // 0..3 (64 cols each)
  const int half = lane >> 4;
  const int l16  = lane & 15;

  const int r0 = tid >> 4;              // staging row stride-16
  const int c0 = (tid & 15) << 1;       // even K offset

  v8f acc[4][4];
#pragma unroll
  for (int i = 0; i < 4; ++i)
#pragma unroll
    for (int j = 0; j < 4; ++j)
      acc[i][j] = (v8f){0.f, 0.f, 0.f, 0.f, 0.f, 0.f, 0.f, 0.f};

  float2 aReg[8], bReg[16];

  auto load_global = [&](int k0) {
#pragma unroll
    for (int it = 0; it < 8; ++it)      // A: 128 rows
      aReg[it] = *(const float2*)(Ag + (size_t)(r0 + it * 16) * HID + k0 + c0);
#pragma unroll
    for (int it = 0; it < 16; ++it)     // B: 256 rows
      bReg[it] = *(const float2*)(Bg + (size_t)(r0 + it * 16) * HID + k0 + c0);
  };
  auto store_lds = [&](int buf) {
#pragma unroll
    for (int it = 0; it < 8; ++it)
      *(unsigned int*)&Als[buf][(r0 + it * 16) * LSTR + c0] =
          pack2_bf16(aReg[it].x, aReg[it].y);
#pragma unroll
    for (int it = 0; it < 16; ++it)
      *(unsigned int*)&Bls[buf][(r0 + it * 16) * LSTR + c0] =
          pack2_bf16(bReg[it].x, bReg[it].y);
  };

  load_global(0);
  store_lds(0);
  __syncthreads();

  const int NK = HID / BK;
  for (int ks = 0; ks < NK; ++ks) {
    const int cur = ks & 1;
    if (ks + 1 < NK) {
      load_global((ks + 1) * BK);                       // in flight during WMMAs
      if (ks + 2 < NK) {
        __builtin_prefetch(Ag + (size_t)r0 * HID + (ks + 2) * BK + c0, 0, 1);
        __builtin_prefetch(Bg + (size_t)r0 * HID + (ks + 2) * BK + c0, 0, 1);
      }
    }

    v16bf afr[4], bfr[4];
#pragma unroll
    for (int mt = 0; mt < 4; ++mt)
      afr[mt] = load_a_frag(Als[cur], wm * 64 + mt * 16, l16, half);
#pragma unroll
    for (int nt = 0; nt < 4; ++nt)
      bfr[nt] = load_b_frag(Bls[cur], wn * 64 + nt * 16, l16, half);

#pragma unroll
    for (int mt = 0; mt < 4; ++mt)
#pragma unroll
      for (int nt = 0; nt < 4; ++nt)
        acc[mt][nt] = __builtin_amdgcn_wmma_f32_16x16x32_bf16(
            false, afr[mt], false, bfr[nt], (short)0, acc[mt][nt], false, false);

    if (ks + 1 < NK) store_lds(cur ^ 1);
    __syncthreads();
  }

  // Epilogue: tanh-GELU, store bf16 h. C layout: VGPR r -> M = r + 8*half, N = l16.
#pragma unroll
  for (int mt = 0; mt < 4; ++mt)
#pragma unroll
    for (int nt = 0; nt < 4; ++nt)
#pragma unroll
      for (int r = 0; r < 8; ++r) {
        int row = tm * K1_BM + wm * 64 + mt * 16 + half * 8 + r;
        int col = tn * K1_BN + wn * 64 + nt * 16 + l16;
        float g = gelu_tanh(acc[mt][nt][r]);
        h[(size_t)(e * TPE + row) * FFN + col] = f32_to_bf16(g);
      }
}

// ---------------- Kernel 2: out = h @ w2  (reduction over FFN) ----------------
// Block tile 128x128, 8 waves in 2x4 grid, wave tile 64x32 (keeps 512 blocks).
#define K2_BM 128
#define K2_BN 128
__global__ __launch_bounds__(256)
void moe_gemm2(const unsigned short* __restrict__ h, const float* __restrict__ w2,
               float* __restrict__ out) {
  __shared__ unsigned short Als[2][K2_BM * LSTR];
  __shared__ unsigned short Bls[2][K2_BN * LSTR];

  const int tid = threadIdx.x;
  const int e   = blockIdx.z;
  const int tm  = blockIdx.y;   // token tile
  const int tn  = blockIdx.x;   // hidden tile

  const unsigned short* Ag = h + (size_t)(e * TPE + tm * K2_BM) * FFN;
  const float* Bg = w2 + (size_t)e * FFN * HID + tn * K2_BN;

  const int wid  = tid >> 5;
  const int lane = tid & 31;
  const int wm   = wid >> 2;
  const int wn   = wid & 3;
  const int half = lane >> 4;
  const int l16  = lane & 15;

  const int r0 = tid >> 4;              // A staging row
  const int c0 = (tid & 15) << 1;       // A even K offset
  const int kB = tid >> 7;              // B staging: k row
  const int nB = tid & 127;             // B staging: n col

  v8f acc[4][2];
#pragma unroll
  for (int i = 0; i < 4; ++i)
#pragma unroll
    for (int j = 0; j < 2; ++j)
      acc[i][j] = (v8f){0.f, 0.f, 0.f, 0.f, 0.f, 0.f, 0.f, 0.f};

  unsigned int aReg[8];
  float bReg[16];

  auto load_global = [&](int k0) {
#pragma unroll
    for (int it = 0; it < 8; ++it)
      aReg[it] = *(const unsigned int*)&Ag[(size_t)(r0 + it * 16) * FFN + k0 + c0];
#pragma unroll
    for (int it = 0; it < 16; ++it)
      bReg[it] = Bg[(size_t)(k0 + kB + it * 2) * HID + nB];   // coalesced along n
  };
  auto store_lds = [&](int buf) {
#pragma unroll
    for (int it = 0; it < 8; ++it)
      *(unsigned int*)&Als[buf][(r0 + it * 16) * LSTR + c0] = aReg[it];
#pragma unroll
    for (int it = 0; it < 16; ++it)
      Bls[buf][nB * LSTR + kB + it * 2] = f32_to_bf16(bReg[it]);  // [n][k]
  };

  load_global(0);
  store_lds(0);
  __syncthreads();

  const int NK = FFN / BK;
  for (int ks = 0; ks < NK; ++ks) {
    const int cur = ks & 1;
    if (ks + 1 < NK) {
      load_global((ks + 1) * BK);
      if (ks + 2 < NK) {
        __builtin_prefetch(Ag + (size_t)r0 * FFN + (ks + 2) * BK + c0, 0, 1);
        __builtin_prefetch(Bg + (size_t)((ks + 2) * BK + kB) * HID + nB, 0, 1);
      }
    }

    v16bf afr[4], bfr[2];
#pragma unroll
    for (int mt = 0; mt < 4; ++mt)
      afr[mt] = load_a_frag(Als[cur], wm * 64 + mt * 16, l16, half);
#pragma unroll
    for (int nt = 0; nt < 2; ++nt)
      bfr[nt] = load_b_frag(Bls[cur], wn * 32 + nt * 16, l16, half);

#pragma unroll
    for (int mt = 0; mt < 4; ++mt)
#pragma unroll
      for (int nt = 0; nt < 2; ++nt)
        acc[mt][nt] = __builtin_amdgcn_wmma_f32_16x16x32_bf16(
            false, afr[mt], false, bfr[nt], (short)0, acc[mt][nt], false, false);

    if (ks + 1 < NK) store_lds(cur ^ 1);
    __syncthreads();
  }

#pragma unroll
  for (int mt = 0; mt < 4; ++mt)
#pragma unroll
    for (int nt = 0; nt < 2; ++nt)
#pragma unroll
      for (int r = 0; r < 8; ++r) {
        int row = tm * K2_BM + wm * 64 + mt * 16 + half * 8 + r;
        int col = tn * K2_BN + wn * 32 + nt * 16 + l16;
        out[(size_t)(e * TPE + row) * HID + col] = acc[mt][nt][r];
      }
}

extern "C" void kernel_launch(void* const* d_in, const int* in_sizes, int n_in,
                              void* d_out, int out_size, void* d_ws, size_t ws_size,
                              hipStream_t stream) {
  (void)in_sizes; (void)n_in; (void)out_size; (void)ws_size;
  const float* x  = (const float*)d_in[0];
  const float* w1 = (const float*)d_in[1];
  const float* w2 = (const float*)d_in[2];
  float* out = (float*)d_out;
  // Workspace: h = gelu(x @ w1^T) as bf16, [8192 x 4096] = 64 MB
  unsigned short* h = (unsigned short*)d_ws;

  dim3 blk(256, 1, 1);
  dim3 g1(FFN / K1_BN, TPE / K1_BM, NEXP);   // 16 x 8 x 8 = 1024 blocks
  hipLaunchKernelGGL(moe_gemm1_gelu, g1, blk, 0, stream, x, w1, h);
  dim3 g2(HID / K2_BN, TPE / K2_BM, NEXP);   // 8 x 8 x 8 = 512 blocks
  hipLaunchKernelGGL(moe_gemm2, g2, blk, 0, stream, h, w2, out);
}
